// MultiHeadAttention_57827439673471
// MI455X (gfx1250) — matrix operand order repaired and checked by reading, not compiled, not run on hardware
//
#include <hip/hip_runtime.h>

// ---------------------------------------------------------------------------
// MultiHeadAttention forward for gfx1250 (MI455X, wave32, WMMA).
// B=2, S=2048, D=1024, H=16, DK=64.
// Pipeline:
//   1) convert f32 -> bf16 (activations + weights)
//   2) gemm_bf16 (mode 0): q/k/v = X @ W^T + b, written head-split bf16
//   3) flash_attn: online-softmax attention, WMMA for q@k^T and p@v
//   4) gemm_bf16 (mode 1): out = attn @ Wo^T + bo, f32 to d_out
// Global->LDS staging uses GLOBAL_LOAD_ASYNC_TO_LDS_B128 (ASYNCcnt) when the
// toolchain exposes the gfx1250 builtin; falls back to VGPR staging otherwise.
// Workspace: 64 MB in d_ws.
// ---------------------------------------------------------------------------

typedef __bf16 bf16;
typedef __attribute__((ext_vector_type(16))) __bf16 v16bf;
typedef __attribute__((ext_vector_type(8)))  float  v8f;

// Builtin-exact pointer types for the gfx1250 async global->LDS loads:
// prototype is (v4i addrspace(1)*, v4i addrspace(3)*, imm offset, imm cpol).
typedef int v4i __attribute__((vector_size(16)));
typedef __attribute__((address_space(1))) v4i gv4i;
typedef __attribute__((address_space(3))) v4i lv4i;

#define B_  2
#define S_  2048
#define D_  1024
#define H_  16
#define DK_ 64

#ifndef __has_builtin
#define __has_builtin(x) 0
#endif
#define HAS_ASYNC_LDS __has_builtin(__builtin_amdgcn_global_load_async_to_lds_b128)

// 16-byte global -> LDS copy. Async (no VGPR staging) when available.
__device__ __forceinline__ void g2lds_b128(const bf16* g, bf16* l) {
#if HAS_ASYNC_LDS
  __builtin_amdgcn_global_load_async_to_lds_b128(
      (gv4i*)g, (lv4i*)l, /*offset=*/0, /*cpol=*/0);
#else
  *(uint4*)l = *(const uint4*)g;
#endif
}

__device__ __forceinline__ void wait_async_all() {
#if HAS_ASYNC_LDS
#if __has_builtin(__builtin_amdgcn_s_wait_asynccnt)
  __builtin_amdgcn_s_wait_asynccnt(0);
#else
  asm volatile("s_wait_asynccnt 0" ::: "memory");
#endif
#endif
}

// D = A(16x32 bf16) x B(32x16 bf16) + C(16x16 f32)
__device__ __forceinline__ v8f wmma_bf16(v16bf a, v16bf b, v8f c) {
  return __builtin_amdgcn_wmma_f32_16x16x32_bf16(
      /*neg_a=*/false, a, /*neg_b=*/false, b,
      /*c_mod=*/(short)0, c, /*reuse_a=*/false, /*reuse_b=*/false);
}

// A-matrix fragment gather (16x32 bf16), ISA 7.12.2 layout:
//   lane L: row = L%16; VGPR0-3 hold K = (L/16)*8 + 0..7,
//                       VGPR4-7 hold K = 16 + (L/16)*8 + 0..7.
__device__ __forceinline__ v16bf load_frag_a(const bf16* __restrict__ p, int ld) {
  const int lane = threadIdx.x & 31;
  const int m = lane & 15;
  const int half = lane >> 4;
  const bf16* row = p + m * ld;
  v16bf f;
#pragma unroll
  for (int i = 0; i < 8; ++i) f[i] = row[half * 8 + i];
#pragma unroll
  for (int i = 0; i < 8; ++i) f[8 + i] = row[16 + half * 8 + i];
  return f;
}

// B-matrix fragment gather (32x16 bf16) from an [N][K] (B^T row-major) tile:
//   lane L: col n = L%16; VGPR0-3 hold K = (L/16)*8 + 0..7,
//                       VGPR4-7 hold K = 16 + (L/16)*8 + 0..7.
__device__ __forceinline__ v16bf load_frag_b(const bf16* __restrict__ p, int ld) {
  const int lane = threadIdx.x & 31;
  const int n = lane & 15;
  const int half = lane >> 4;
  const bf16* row = p + n * ld;
  v16bf f;
#pragma unroll
  for (int i = 0; i < 8; ++i) f[i] = row[half * 8 + i];
#pragma unroll
  for (int i = 0; i < 8; ++i) f[8 + i] = row[16 + half * 8 + i];
  return f;
}

// ---------------------------------------------------------------------------
__global__ void cvt_f32_to_bf16(const float* __restrict__ x,
                                bf16* __restrict__ y, int n) {
  int i = blockIdx.x * blockDim.x + threadIdx.x;
  if (i < n) y[i] = (bf16)x[i];
}

// ---------------------------------------------------------------------------
// C[M=4096][N=1024] = A[M][1024] @ W[N][1024]^T + bias.
// Block: 256 threads = 8 wave32 waves; tile BM=128, BN=128, BK=32.
// Wave (wm 0..3, wn 0..1) owns a 32x64 sub-tile = 2x4 WMMA frags = 8 WMMAs
// per K-step (doubles matrix-op density per barrier vs 32x32 waves).
// mode 0: bf16 output, head-split [B,H,S,DK]; mode 1: f32 output [M][N].
__global__ __launch_bounds__(256) void gemm_bf16(
    const bf16* __restrict__ A, const bf16* __restrict__ W,
    const float* __restrict__ bias,
    bf16* __restrict__ out_split, float* __restrict__ out_f32, int mode) {
  __shared__ __attribute__((aligned(16))) bf16 As[128 * 32];
  __shared__ __attribute__((aligned(16))) bf16 Bs[128 * 32];

  const int tid = threadIdx.x;
  const int lane = tid & 31;
  const int wid = tid >> 5;
  const int wm = wid >> 1, wn = wid & 1;
  const int bm = blockIdx.x * 128, bn = blockIdx.y * 128;

  v8f acc[2][4] = {};

  for (int k0 = 0; k0 < D_; k0 += 32) {
    // Stage A and B tiles: 2 x (128x32 bf16) = 2 x 512 x 16B chunks.
#pragma unroll
    for (int i = 0; i < 2; ++i) {
      int c = tid + i * 256;
      int row = c >> 2, q = c & 3;
      g2lds_b128(&A[(size_t)(bm + row) * D_ + k0 + q * 8],
                 &As[row * 32 + q * 8]);
      g2lds_b128(&W[(size_t)(bn + row) * D_ + k0 + q * 8],
                 &Bs[row * 32 + q * 8]);
    }
    // Hint the next A tile toward L2 (global_prefetch_b8).
    __builtin_prefetch(&A[(size_t)(bm + (tid >> 2)) * D_ + k0 + 32], 0, 1);
    wait_async_all();
    __syncthreads();

    v16bf a0 = load_frag_a(&As[(wm * 32) * 32], 32);
    v16bf a1 = load_frag_a(&As[(wm * 32 + 16) * 32], 32);
#pragma unroll
    for (int ni = 0; ni < 4; ++ni) {
      v16bf b = load_frag_b(&Bs[(wn * 64 + ni * 16) * 32], 32);
      acc[0][ni] = wmma_bf16(a0, b, acc[0][ni]);
      acc[1][ni] = wmma_bf16(a1, b, acc[1][ni]);
    }
    __syncthreads();
  }

  // Epilogue. C layout: reg r, lane L -> row r + 8*(L/16), col L%16.
  const int half = lane >> 4, col0 = lane & 15;
#pragma unroll
  for (int mi = 0; mi < 2; ++mi)
#pragma unroll
    for (int ni = 0; ni < 4; ++ni)
#pragma unroll
      for (int r = 0; r < 8; ++r) {
        int m = bm + wm * 32 + mi * 16 + r + 8 * half;
        int n = bn + wn * 64 + ni * 16 + col0;
        float v = acc[mi][ni][r] + bias[n];
        if (mode == 0) {
          int b = m >> 11, s = m & (S_ - 1);
          int h = n >> 6, dk = n & 63;
          out_split[(((size_t)(b * H_ + h)) * S_ + s) * DK_ + dk] = (bf16)v;
        } else {
          out_f32[(size_t)m * D_ + n] = v;
        }
      }
}

// ---------------------------------------------------------------------------
// Flash attention: one block per (b*H+h, 64-row q tile). 128 threads = 4
// waves; wave w owns q rows [w*16, w*16+16) and streams 64-key tiles with
// online softmax. Causal mask applied analytically (reference mask is tril).
__global__ __launch_bounds__(128) void flash_attn(
    const bf16* __restrict__ qh, const bf16* __restrict__ kh,
    const bf16* __restrict__ vh, bf16* __restrict__ attn) {
  __shared__ __attribute__((aligned(16))) bf16 Qs[64 * 64];
  __shared__ __attribute__((aligned(16))) bf16 Ks[64 * 64];        // [s'][dk]
  __shared__ __attribute__((aligned(16))) bf16 Vt[64 * 64];        // [dk][s']
  __shared__ __attribute__((aligned(16))) bf16 Ps[4][16 * 64];     // per-wave P

  const int tid = threadIdx.x, lane = tid & 31, w = tid >> 5;
  const int qt = blockIdx.x & 31;    // 32 q tiles per head-slice
  const int bh = blockIdx.x >> 5;    // b*H + h
  const size_t base = (size_t)bh * S_ * DK_;
  const bf16* qptr = qh + base + (size_t)qt * 64 * DK_;

#pragma unroll
  for (int i = 0; i < 4; ++i) {
    int c = tid + i * 128;
    g2lds_b128(&qptr[c * 8], &Qs[c * 8]);
  }
  wait_async_all();
  __syncthreads();

  const int half = lane >> 4, col0 = lane & 15;
  v16bf qa0 = load_frag_a(&Qs[(w * 16) * 64 + 0], 64);
  v16bf qa1 = load_frag_a(&Qs[(w * 16) * 64 + 32], 64);

  float m_run[8], l_run[8];
  v8f o[4] = {};
#pragma unroll
  for (int r = 0; r < 8; ++r) { m_run[r] = -3.0e38f; l_run[r] = 0.0f; }

  for (int jt = 0; jt <= qt; ++jt) {
    const bf16* kptr = kh + base + (size_t)jt * 64 * DK_;
    const bf16* vptr = vh + base + (size_t)jt * 64 * DK_;
#pragma unroll
    for (int i = 0; i < 4; ++i) {
      int c = tid + i * 128;
      g2lds_b128(&kptr[c * 8], &Ks[c * 8]);
    }
    // V transposed into LDS so p@v's B operand is [n=dk][k=s'].
#pragma unroll
    for (int i = 0; i < 32; ++i) {
      int idx = i * 128 + tid;
      int sp = idx >> 6, dk = idx & 63;
      Vt[dk * 64 + sp] = vptr[idx];
    }
    wait_async_all();
    __syncthreads();

    // Scores: 16x64 per wave, 2 K-steps of WMMA per 16-key column tile.
    v8f st[4];
#pragma unroll
    for (int t = 0; t < 4; ++t) {
      v16bf kb0 = load_frag_b(&Ks[(t * 16) * 64 + 0], 64);
      v16bf kb1 = load_frag_b(&Ks[(t * 16) * 64 + 32], 64);
      v8f s = {};
      s = wmma_bf16(qa0, kb0, s);
      s = wmma_bf16(qa1, kb1, s);
      st[t] = s;
    }

    const float scale = 0.125f;  // 1/sqrt(64)
    const bool diag = (jt == qt);
#pragma unroll
    for (int t = 0; t < 4; ++t)
#pragma unroll
      for (int r = 0; r < 8; ++r) {
        float v = st[t][r] * scale;
        if (diag) {
          int row_l = w * 16 + r + 8 * half;
          int col_l = t * 16 + col0;
          if (col_l > row_l) v = -3.0e38f;
        }
        st[t][r] = v;
      }

    // Per-row max across 16 lanes of each wave half (rows are half-local).
    float mnew[8];
#pragma unroll
    for (int r = 0; r < 8; ++r) {
      float v = fmaxf(fmaxf(st[0][r], st[1][r]), fmaxf(st[2][r], st[3][r]));
      v = fmaxf(v, __shfl_xor(v, 1, 32));
      v = fmaxf(v, __shfl_xor(v, 2, 32));
      v = fmaxf(v, __shfl_xor(v, 4, 32));
      v = fmaxf(v, __shfl_xor(v, 8, 32));
      mnew[r] = fmaxf(m_run[r], v);
    }

    // P = exp(S - m_new); stage bf16 P through LDS to re-gather as A frags.
#pragma unroll
    for (int t = 0; t < 4; ++t)
#pragma unroll
      for (int r = 0; r < 8; ++r) {
        float p = __expf(st[t][r] - mnew[r]);
        st[t][r] = p;
        Ps[w][(r + 8 * half) * 64 + t * 16 + col0] = (bf16)p;
      }

#pragma unroll
    for (int r = 0; r < 8; ++r) {
      float rs = st[0][r] + st[1][r] + st[2][r] + st[3][r];
      rs += __shfl_xor(rs, 1, 32);
      rs += __shfl_xor(rs, 2, 32);
      rs += __shfl_xor(rs, 4, 32);
      rs += __shfl_xor(rs, 8, 32);
      float corr = __expf(m_run[r] - mnew[r]);
      l_run[r] = l_run[r] * corr + rs;
      m_run[r] = mnew[r];
#pragma unroll
      for (int t = 0; t < 4; ++t) o[t][r] *= corr;
    }

    // CDNA5 split wait counter: LDS stores above must land before re-gather.
    asm volatile("s_wait_dscnt 0" ::: "memory");

    v16bf pa0 = load_frag_a(&Ps[w][0], 64);
    v16bf pa1 = load_frag_a(&Ps[w][32], 64);
#pragma unroll
    for (int t = 0; t < 4; ++t) {
      v16bf vb0 = load_frag_b(&Vt[(t * 16) * 64 + 0], 64);
      v16bf vb1 = load_frag_b(&Vt[(t * 16) * 64 + 32], 64);
      o[t] = wmma_bf16(pa0, vb0, o[t]);
      o[t] = wmma_bf16(pa1, vb1, o[t]);
    }
    __syncthreads();  // before Ks/Vt are overwritten
  }

  // O /= l ; write to attn[B,S,D] (heads re-concatenated) as bf16.
  const int b = bh >> 4, h = bh & 15;
#pragma unroll
  for (int t = 0; t < 4; ++t)
#pragma unroll
    for (int r = 0; r < 8; ++r) {
      int srow = qt * 64 + w * 16 + r + 8 * half;
      int dk = t * 16 + col0;
      float val = o[t][r] / l_run[r];
      attn[((size_t)(b * S_ + srow)) * D_ + h * 64 + dk] = (bf16)val;
    }
}

// ---------------------------------------------------------------------------
extern "C" void kernel_launch(void* const* d_in, const int* in_sizes, int n_in,
                              void* d_out, int out_size, void* d_ws, size_t ws_size,
                              hipStream_t stream) {
  const float* Q  = (const float*)d_in[0];
  const float* K  = (const float*)d_in[1];
  const float* V  = (const float*)d_in[2];
  // d_in[3] = causal mask; applied analytically inside flash_attn.
  const float* Wq = (const float*)d_in[4];
  const float* bq = (const float*)d_in[5];
  const float* Wk = (const float*)d_in[6];
  const float* bk = (const float*)d_in[7];
  const float* Wv = (const float*)d_in[8];
  const float* bv = (const float*)d_in[9];
  const float* Wo = (const float*)d_in[10];
  const float* bo = (const float*)d_in[11];
  float* out = (float*)d_out;

  const size_t NSD = (size_t)B_ * S_ * D_;  // 4M activation elements
  const size_t NDD = (size_t)D_ * D_;       // 1M weight elements

  char* p = (char*)d_ws;
  bf16* Qb  = (bf16*)p; p += NSD * 2;
  bf16* Kb  = (bf16*)p; p += NSD * 2;
  bf16* Vb  = (bf16*)p; p += NSD * 2;
  bf16* Wqb = (bf16*)p; p += NDD * 2;
  bf16* Wkb = (bf16*)p; p += NDD * 2;
  bf16* Wvb = (bf16*)p; p += NDD * 2;
  bf16* Wob = (bf16*)p; p += NDD * 2;
  bf16* qhd = (bf16*)p; p += NSD * 2;
  bf16* khd = (bf16*)p; p += NSD * 2;
  bf16* vhd = (bf16*)p; p += NSD * 2;
  bf16* att = (bf16*)p; p += NSD * 2;

  // 1) f32 -> bf16 converts.
  cvt_f32_to_bf16<<<(int)(NSD / 256), 256, 0, stream>>>(Q, Qb, (int)NSD);
  cvt_f32_to_bf16<<<(int)(NSD / 256), 256, 0, stream>>>(K, Kb, (int)NSD);
  cvt_f32_to_bf16<<<(int)(NSD / 256), 256, 0, stream>>>(V, Vb, (int)NSD);
  cvt_f32_to_bf16<<<(int)(NDD / 256), 256, 0, stream>>>(Wq, Wqb, (int)NDD);
  cvt_f32_to_bf16<<<(int)(NDD / 256), 256, 0, stream>>>(Wk, Wkb, (int)NDD);
  cvt_f32_to_bf16<<<(int)(NDD / 256), 256, 0, stream>>>(Wv, Wvb, (int)NDD);
  cvt_f32_to_bf16<<<(int)(NDD / 256), 256, 0, stream>>>(Wo, Wob, (int)NDD);

  // 2) Projections -> head-split bf16.
  dim3 gg(B_ * S_ / 128, D_ / 128);
  gemm_bf16<<<gg, 256, 0, stream>>>(Qb, Wqb, bq, qhd, nullptr, 0);
  gemm_bf16<<<gg, 256, 0, stream>>>(Kb, Wkb, bk, khd, nullptr, 0);
  gemm_bf16<<<gg, 256, 0, stream>>>(Vb, Wvb, bv, vhd, nullptr, 0);

  // 3) Flash attention: B*H * (S/64) blocks.
  flash_attn<<<B_ * H_ * (S_ / 64), 128, 0, stream>>>(qhd, khd, vhd, att);

  // 4) Output projection, f32 result to d_out.
  gemm_bf16<<<gg, 256, 0, stream>>>(att, Wob, bo, nullptr, out, 1);
}
